// CustomMultiheadAttention_24704651887211
// MI455X (gfx1250) — compile-verified, matrix operand order checked by reading, not run on hardware
//
#include <hip/hip_runtime.h>

// ---------------------------------------------------------------- types
typedef __bf16 bf16;
typedef float v8f  __attribute__((ext_vector_type(8)));
typedef bf16  v16bf __attribute__((ext_vector_type(16)));
typedef bf16  v8bf  __attribute__((ext_vector_type(8)));

#define EMBED 1024
#define RANK  512
#define HEADS 16
#define HDIM  64
#define BATCH 4
#define SEQ   8192
#define MROWS (BATCH * SEQ)

// ---------------------------------------------------------------- helpers
__device__ inline v8bf ld8(const bf16* p) { return *(const v8bf*)p; }
__device__ inline v8bf ld8(const float* p) {
  v8f x = *(const v8f*)p;
  v8bf r;
#pragma unroll
  for (int i = 0; i < 8; ++i) r[i] = (bf16)x[i];
  return r;
}

// Per-lane A/B fragment for v_wmma_f32_16x16x32_bf16 from a K-contiguous row.
// lane holds row (lane%16); chunk0 at k=(lane/16)*8, chunk1 at +16.
template <typename T>
__device__ inline v16bf frag16(const T* p) {
  v8bf lo = ld8(p);
  v8bf hi = ld8(p + 16);
  return __builtin_shufflevector(lo, hi, 0, 1, 2, 3, 4, 5, 6, 7, 8, 9, 10, 11,
                                 12, 13, 14, 15);
}

__device__ inline v8f wmma_bf16(v16bf a, v16bf b, v8f c) {
  return __builtin_amdgcn_wmma_f32_16x16x32_bf16(false, a, false, b, (short)0,
                                                 c, false, false);
}

// Async copy of 16 bytes global -> LDS (per lane), tracked with ASYNCcnt.
__device__ inline void async_copy_b128(const void* gptr, void* lds_ptr) {
  unsigned lds_off = (unsigned)(size_t)lds_ptr;  // flat addr low 32b = LDS offset
  asm volatile("global_load_async_to_lds_b128 %0, %1, off"
               :
               : "v"(lds_off), "v"(gptr)
               : "memory");
}
__device__ inline void async_wait0() {
  asm volatile("s_wait_asynccnt 0x0" ::: "memory");
}

// ---------------------------------------------------------------- small kernels
__global__ void cvt_f32_bf16(const float* __restrict__ s, bf16* __restrict__ d,
                             int n) {
  int i = blockIdx.x * blockDim.x + threadIdx.x;
  if (i < n) d[i] = (bf16)s[i];
}

__global__ void zero_f32(float* __restrict__ p, int n) {
  int i = blockIdx.x * blockDim.x + threadIdx.x;
  if (i < n) p[i] = 0.0f;
}

// ---------------------------------------------------------------- generic GEMM
// C = act(A[M,K] @ W[N,K].T + bias). Wave tile 32x64 (two 16-row subtiles
// sharing B fragments), 4 waves/block -> block tile 128x64. Software-pipelined:
// fragments for k+32 are loaded before the k-step WMMAs; sched_barrier pins
// the load clause ahead of the WMMA burst so the post-RA scheduler cannot
// sink loads back to their consumers (which would force s_wait_loadcnt 0).
template <typename TA, int ACT /*0 none, 1 elu+1*/, bool BIAS, bool OUT32>
__global__ __launch_bounds__(128) void gemm_xwt(const TA* __restrict__ A,
                                                const bf16* __restrict__ W,
                                                const float* __restrict__ bias,
                                                void* __restrict__ Cout, int M,
                                                int Nout, int K) {
  const int lane = threadIdx.x & 31;
  const int wid = threadIdx.x >> 5;
  const int lm = lane & 15;
  const int half_id = lane >> 4;  // 0/1
  const int klo = half_id * 8;

  const int mbase = (blockIdx.y * 4 + wid) * 32;
  const int nbase = blockIdx.x * 64;

  const TA* arow0 = A + (size_t)(mbase + lm) * K + klo;
  const TA* arow1 = A + (size_t)(mbase + 16 + lm) * K + klo;
  const bf16* w0 = W + (size_t)(nbase + lm) * K + klo;

  v8f acc0[4] = {v8f{}, v8f{}, v8f{}, v8f{}};
  v8f acc1[4] = {v8f{}, v8f{}, v8f{}, v8f{}};

  // prologue: fragments for k = 0
  v16bf a0 = frag16(arow0);
  v16bf a1 = frag16(arow1);
  v16bf b[4];
#pragma unroll
  for (int j = 0; j < 4; ++j) b[j] = frag16(w0 + (size_t)(j * 16) * K);

  for (int k = 32; k < K; k += 32) {
    // prefetch next iteration's fragments (issued before this step's WMMAs)
    v16bf a0n = frag16(arow0 + k);
    v16bf a1n = frag16(arow1 + k);
    v16bf bn[4];
#pragma unroll
    for (int j = 0; j < 4; ++j) bn[j] = frag16(w0 + (size_t)(j * 16) * K + k);

    __builtin_amdgcn_sched_barrier(0);  // keep loads above, WMMAs below

#pragma unroll
    for (int j = 0; j < 4; ++j) {
      acc0[j] = wmma_bf16(a0, b[j], acc0[j]);
      acc1[j] = wmma_bf16(a1, b[j], acc1[j]);
    }
    a0 = a0n;
    a1 = a1n;
#pragma unroll
    for (int j = 0; j < 4; ++j) b[j] = bn[j];
  }
  // drain last step
#pragma unroll
  for (int j = 0; j < 4; ++j) {
    acc0[j] = wmma_bf16(a0, b[j], acc0[j]);
    acc1[j] = wmma_bf16(a1, b[j], acc1[j]);
  }

#pragma unroll
  for (int j = 0; j < 4; ++j) {
    const int col = nbase + j * 16 + lm;
    float bv = BIAS ? bias[col] : 0.0f;
#pragma unroll
    for (int s = 0; s < 2; ++s) {
      v8f* accp = s ? &acc1[j] : &acc0[j];
#pragma unroll
      for (int r = 0; r < 8; ++r) {
        float x = (*accp)[r] + bv;
        if (ACT == 1) x = (x > 0.0f) ? (x + 1.0f) : __expf(x);  // elu(x)+1
        const size_t row = (size_t)(mbase + s * 16 + r + 8 * half_id);
        if (OUT32) {
          // final result: written once, never re-read -> non-temporal store
          __builtin_nontemporal_store(x, &((float*)Cout)[row * Nout + col]);
        } else {
          ((bf16*)Cout)[row * Nout + col] = (bf16)x;
        }
      }
    }
  }
}

// ---------------------------------------------------------------- phase 1
// Per (b,h): kvAcc[e][d] += sum_n k[n,d]*v[n,e]  (stored transposed -> kvT),
// ksum[d] += sum_n k[n,d]. Sequence split 8 ways, fp32 global atomics.
// K/V tiles staged into LDS with gfx1250 async global->LDS copies.
__global__ __launch_bounds__(256) void attn_phase1(const bf16* __restrict__ kp,
                                                   const bf16* __restrict__ vp,
                                                   float* __restrict__ kvAcc,
                                                   float* __restrict__ ksAcc) {
  const int h = blockIdx.x, b = blockIdx.y, seg = blockIdx.z;
  const int t = threadIdx.x;
  __shared__ __align__(16) bf16 ksh[32 * 64];
  __shared__ __align__(16) bf16 vsh[32 * 64];

  const int d0 = (t & 15) * 4;
  const int e0 = (t >> 4) * 4;
  const int lrow = t >> 3;
  const int lcol = (t & 7) * 8;

  float acc[4][4] = {};
  float ks = 0.0f;

  const size_t base = ((size_t)b * SEQ + (size_t)seg * (SEQ / 8)) * EMBED +
                      (size_t)h * HDIM;

  for (int c = 0; c < SEQ / 8; c += 32) {
    // async DMA-style staging: 16B per lane, tracked by ASYNCcnt
    async_copy_b128(kp + base + (size_t)(c + lrow) * EMBED + lcol,
                    &ksh[lrow * 64 + lcol]);
    async_copy_b128(vp + base + (size_t)(c + lrow) * EMBED + lcol,
                    &vsh[lrow * 64 + lcol]);
    async_wait0();
    __syncthreads();
#pragma unroll 4
    for (int n = 0; n < 32; ++n) {
      float kj[4], vi[4];
#pragma unroll
      for (int j = 0; j < 4; ++j) kj[j] = (float)ksh[n * 64 + d0 + j];
#pragma unroll
      for (int i = 0; i < 4; ++i) vi[i] = (float)vsh[n * 64 + e0 + i];
#pragma unroll
      for (int i = 0; i < 4; ++i)
#pragma unroll
        for (int j = 0; j < 4; ++j) acc[i][j] += vi[i] * kj[j];
    }
    if (t < 64) {
      for (int n = 0; n < 32; ++n) ks += (float)ksh[n * 64 + t];
    }
    __syncthreads();
  }

  float* kva = kvAcc + (size_t)(b * HEADS + h) * HDIM * HDIM;
#pragma unroll
  for (int i = 0; i < 4; ++i)
#pragma unroll
    for (int j = 0; j < 4; ++j)
      atomicAdd(&kva[(e0 + i) * HDIM + d0 + j], acc[i][j]);  // kvT layout [e][d]
  if (t < 64) atomicAdd(&ksAcc[(b * HEADS + h) * HDIM + t], ks);
}

// ---------------------------------------------------------------- phase 2
// num = Q @ kv (WMMA, kv supplied as kvT rows), denom = Q . ksum,
// attn = num / (denom + 1e-6) written IN PLACE over qp (bf16).
// All 10 fragments loaded first (pinned by sched_barrier), then an 8-WMMA
// burst, then the denominator; loads overlap the matrix pipe.
__global__ __launch_bounds__(128) void attn_phase2(bf16* __restrict__ qp,
                                                   const bf16* __restrict__ kvT,
                                                   const float* __restrict__ ksum) {
  const int lane = threadIdx.x & 31;
  const int wid = threadIdx.x >> 5;
  const int lm = lane & 15;
  const int half_id = lane >> 4;
  const int klo = half_id * 8;

  const int b = blockIdx.z, h = blockIdx.y;
  const int n0 = (blockIdx.x * 4 + wid) * 16;
  const size_t row0 = (size_t)b * SEQ + n0;

  const bf16* kv = kvT + (size_t)(b * HEADS + h) * HDIM * HDIM;
  const float* ks = ksum + (size_t)(b * HEADS + h) * HDIM;

  const bf16* qrow = qp + (row0 + lm) * EMBED + (size_t)h * HDIM;

  // hoist ALL fragment loads (A for both k-steps + 8 B fragments)
  v16bf a[2];
  v16bf bb[8];
#pragma unroll
  for (int kk = 0; kk < 2; ++kk) {
    a[kk] = frag16(qrow + kk * 32 + klo);
#pragma unroll
    for (int j = 0; j < 4; ++j)
      bb[kk * 4 + j] = frag16(kv + (size_t)(j * 16 + lm) * HDIM + kk * 32 + klo);
  }

  __builtin_amdgcn_sched_barrier(0);  // loads above, WMMA burst below

  v8f acc[4] = {v8f{}, v8f{}, v8f{}, v8f{}};
#pragma unroll
  for (int kk = 0; kk < 2; ++kk)
#pragma unroll
    for (int j = 0; j < 4; ++j) acc[j] = wmma_bf16(a[kk], bb[kk * 4 + j], acc[j]);

  __builtin_amdgcn_sched_barrier(0);  // keep denominator work after the burst

  // denominator for row (lane%16)
  float dn = 0.0f;
#pragma unroll
  for (int d = 0; d < HDIM; ++d) dn += (float)qrow[d] * ks[d];

#pragma unroll
  for (int r = 0; r < 8; ++r) {
    float dm = __shfl(dn, r + 8 * half_id, 32);  // denom of absolute row r+8*half
    float rc = 1.0f / (dm + 1e-6f);
    const size_t row = row0 + r + 8 * half_id;
#pragma unroll
    for (int j = 0; j < 4; ++j)
      qp[row * EMBED + (size_t)h * HDIM + j * 16 + lm] = (bf16)(acc[j][r] * rc);
  }
}

// ---------------------------------------------------------------- launch
extern "C" void kernel_launch(void* const* d_in, const int* in_sizes, int n_in,
                              void* d_out, int out_size, void* d_ws,
                              size_t ws_size, hipStream_t stream) {
  const float* query = (const float*)d_in[0];
  const float* key_ = (const float*)d_in[1];
  const float* value = (const float*)d_in[2];
  const float* Wq_down = (const float*)d_in[3];
  const float* Wq_up = (const float*)d_in[4];
  const float* bq_up = (const float*)d_in[5];
  const float* Wk_down = (const float*)d_in[6];
  const float* Wk_up = (const float*)d_in[7];
  const float* bk_up = (const float*)d_in[8];
  const float* Wv_down = (const float*)d_in[9];
  const float* Wv_up = (const float*)d_in[10];
  const float* bv_up = (const float*)d_in[11];
  const float* Wo = (const float*)d_in[12];
  const float* bo = (const float*)d_in[13];
  float* out = (float*)d_out;

  // workspace carve-up (256B aligned)
  char* w = (char*)d_ws;
  auto carve = [&](size_t bytes) {
    char* p = w;
    w += (bytes + 255) & ~(size_t)255;
    return p;
  };
  bf16* wqd = (bf16*)carve((size_t)RANK * EMBED * 2);
  bf16* wqu = (bf16*)carve((size_t)EMBED * RANK * 2);
  bf16* wkd = (bf16*)carve((size_t)RANK * EMBED * 2);
  bf16* wku = (bf16*)carve((size_t)EMBED * RANK * 2);
  bf16* wvd = (bf16*)carve((size_t)RANK * EMBED * 2);
  bf16* wvu = (bf16*)carve((size_t)EMBED * RANK * 2);
  bf16* wo = (bf16*)carve((size_t)EMBED * EMBED * 2);
  bf16* tbuf = (bf16*)carve((size_t)MROWS * RANK * 2);
  bf16* qp = (bf16*)carve((size_t)MROWS * EMBED * 2);
  bf16* kp = (bf16*)carve((size_t)MROWS * EMBED * 2);
  bf16* vp = (bf16*)carve((size_t)MROWS * EMBED * 2);
  float* kvAcc = (float*)carve((size_t)BATCH * HEADS * HDIM * HDIM * 4);
  bf16* kvT = (bf16*)carve((size_t)BATCH * HEADS * HDIM * HDIM * 2);
  float* ksAcc = (float*)carve((size_t)BATCH * HEADS * HDIM * 4);

  // 1. convert weights to bf16
  auto cvt = [&](const float* s, bf16* d, int n) {
    cvt_f32_bf16<<<(n + 255) / 256, 256, 0, stream>>>(s, d, n);
  };
  cvt(Wq_down, wqd, RANK * EMBED);
  cvt(Wq_up, wqu, EMBED * RANK);
  cvt(Wk_down, wkd, RANK * EMBED);
  cvt(Wk_up, wku, EMBED * RANK);
  cvt(Wv_down, wvd, RANK * EMBED);
  cvt(Wv_up, wvu, EMBED * RANK);
  cvt(Wo, wo, EMBED * EMBED);

  const dim3 blk(128);
  const dim3 gDown(RANK / 64, MROWS / 128);
  const dim3 gUp(EMBED / 64, MROWS / 128);

  // 2. Q path: down (fp32 A), up (+bias, elu+1)
  gemm_xwt<float, 0, false, false>
      <<<gDown, blk, 0, stream>>>(query, wqd, nullptr, tbuf, MROWS, RANK, EMBED);
  gemm_xwt<bf16, 1, true, false>
      <<<gUp, blk, 0, stream>>>(tbuf, wqu, bq_up, qp, MROWS, EMBED, RANK);
  // 3. K path
  gemm_xwt<float, 0, false, false>
      <<<gDown, blk, 0, stream>>>(key_, wkd, nullptr, tbuf, MROWS, RANK, EMBED);
  gemm_xwt<bf16, 1, true, false>
      <<<gUp, blk, 0, stream>>>(tbuf, wku, bk_up, kp, MROWS, EMBED, RANK);
  // 4. V path (no activation)
  gemm_xwt<float, 0, false, false>
      <<<gDown, blk, 0, stream>>>(value, wvd, nullptr, tbuf, MROWS, RANK, EMBED);
  gemm_xwt<bf16, 0, true, false>
      <<<gUp, blk, 0, stream>>>(tbuf, wvu, bv_up, vp, MROWS, EMBED, RANK);

  // 5. phase 1: kv^T and ksum (zero kvAcc first — ws is poisoned)
  {
    int nz = BATCH * HEADS * HDIM * HDIM + BATCH * HEADS * HDIM;
    zero_f32<<<(nz + 255) / 256, 256, 0, stream>>>(kvAcc, nz);  // zeroes kvAcc and first 16KB of kvT; ksAcc not zeroed
    attn_phase1<<<dim3(HEADS, BATCH, 8), 256, 0, stream>>>(kp, vp, kvAcc, ksAcc);
    int nk = BATCH * HEADS * HDIM * HDIM;
    cvt_f32_bf16<<<(nk + 255) / 256, 256, 0, stream>>>(kvAcc, kvT, nk);
  }

  // 6. phase 2: num/denom, normalize, attn written in place over qp
  attn_phase2<<<dim3(SEQ / 64, HEADS, BATCH), 128, 0, stream>>>(qp, kvT, ksAcc);

  // 7. output projection (fp32 result)
  gemm_xwt<bf16, 0, true, true>
      <<<gUp, blk, 0, stream>>>(qp, wo, bo, out, MROWS, EMBED, EMBED);
}
